// HybridSlotQueryModel_40269613367727
// MI455X (gfx1250) — matrix operand order repaired and checked by reading, compiled, not run on hardware
//
#include <hip/hip_runtime.h>
#include <hip/hip_bf16.h>
#include <math.h>

// ---------------------------------------------------------------------------
// Model dims
#define BDIM 8
#define TDIM 16
#define NDIM 256
#define DDIM 1280
#define HDIM 512
#define SDIM 3
#define NITER 3
#define NHEADS 8
#define HD 64
#define ROWS (BDIM * TDIM * NDIM)   // 32768

typedef __attribute__((ext_vector_type(16))) __bf16 bf16x16;
typedef __attribute__((ext_vector_type(8)))  float  f32x8;

__device__ __forceinline__ f32x8 wmma_bf16(bf16x16 a, bf16x16 b, f32x8 c) {
  return __builtin_amdgcn_wmma_f32_16x16x32_bf16(false, a, false, b, (short)0, c,
                                                 false, false);
}

// A fragment (16x32 bf16, MxK) gathered from a row-major bf16 buffer.
// Lanes 0-15: row M=lane, K in {0..7, 16..23}; lanes 16-31: M=lane-16, K in {8..15, 24..31}.
__device__ __forceinline__ bf16x16 a_frag(const __bf16* base, int stride, int lane, int k) {
  int row = lane & 15;
  int kb  = (lane >> 4) << 3;  // 0 or 8
  const __bf16* p = base + row * stride + k + kb;
  bf16x16 r;
#pragma unroll
  for (int i = 0; i < 8; ++i) r[i] = p[i];
#pragma unroll
  for (int i = 0; i < 8; ++i) r[8 + i] = p[16 + i];
  return r;
}

// B fragment (32x16 bf16, KxN) from a pre-converted row-major bf16 weight [out, ldw]:
// col = colbase + (lane&15); lanes<16: K {0..7, 16..23}, else K {8..15, 24..31}.
__device__ __forceinline__ bf16x16 b_frag_bf16(const __bf16* w, int ldw, int colbase,
                                               int k, int lane) {
  const __bf16* p = w + (size_t)(colbase + (lane & 15)) * ldw + k + ((lane >> 4) << 3);
  bf16x16 r;
#pragma unroll
  for (int i = 0; i < 8; ++i) r[i] = p[i];
#pragma unroll
  for (int i = 0; i < 8; ++i) r[8 + i] = p[16 + i];
  return r;
}

__device__ __forceinline__ f32x8 zero8() {
  f32x8 z;
#pragma unroll
  for (int i = 0; i < 8; ++i) z[i] = 0.0f;
  return z;
}

// ---------------------------------------------------------------------------
// Kernel W: one-time f32 -> bf16 weight conversion (weights then live in L2)
__global__ __launch_bounds__(256) void kW_cvt(const float* __restrict__ src,
                                              __bf16* __restrict__ dst, int n) {
  int i = blockIdx.x * 256 + threadIdx.x;
  if (i < n) dst[i] = (__bf16)src[i];
}

// ---------------------------------------------------------------------------
// Kernel A: per 16-row tile over all B*T*N rows:
//   y1 = LN(X @ w_in^T + b_in)                 (f32, workspace)
//   hm = relu((Xnext - X) @ w_m1p^T + b_m1p)   (bf16, workspace)
__global__ __launch_bounds__(256) void kA_proj(
    const float* __restrict__ X,
    const __bf16* __restrict__ w_in_bf, const float* __restrict__ b_in,
    const float* __restrict__ g_ln_in, const float* __restrict__ b_ln_in,
    const __bf16* __restrict__ w_m1p_bf, const float* __restrict__ b_m1p,
    float* __restrict__ y1, __bf16* __restrict__ hm) {
  __shared__ __bf16 tx[16 * 32];
  __shared__ __bf16 tm[16 * 32];
  __shared__ float rsum[16], rsq[16];

  const int tid = threadIdx.x;
  const int lane = tid & 31;
  const int wave = tid >> 5;
  const int row0 = blockIdx.x * 16;
  const int t = (row0 / NDIM) % TDIM;
  const long long xbase = (long long)row0 * DDIM;
  const long long xnext = (t < TDIM - 1) ? xbase + (long long)NDIM * DDIM : xbase;

  f32x8 acc1[4], acc2[4];
#pragma unroll
  for (int c = 0; c < 4; ++c) { acc1[c] = zero8(); acc2[c] = zero8(); }

  for (int k = 0; k < DDIM; k += 32) {
    // cooperative tile load: 16 rows x 32 cols, 2 elems/thread
    {
      int e = tid * 2;
      int r = e >> 5, c = e & 31;
      long long o = (long long)r * DDIM + k + c;
      float x0 = X[xbase + o], x1 = X[xbase + o + 1];
      float n0 = X[xnext + o], n1 = X[xnext + o + 1];
      tx[r * 32 + c] = (__bf16)x0;
      tx[r * 32 + c + 1] = (__bf16)x1;
      tm[r * 32 + c] = (__bf16)(n0 - x0);
      tm[r * 32 + c + 1] = (__bf16)(n1 - x1);
      if (k + 32 < DDIM) {  // pull next k-step toward the WGP (global_prefetch_b8)
        __builtin_prefetch(&X[xbase + o + 32], 0, 0);
        __builtin_prefetch(&X[xnext + o + 32], 0, 0);
      }
    }
    __syncthreads();
    bf16x16 ax = a_frag(tx, 32, lane, 0);
    bf16x16 am = a_frag(tm, 32, lane, 0);
    __syncthreads();
#pragma unroll
    for (int ct = 0; ct < 4; ++ct) {
      int cb = wave * 64 + ct * 16;
      bf16x16 b1 = b_frag_bf16(w_in_bf, DDIM, cb, k, lane);
      acc1[ct] = wmma_bf16(ax, b1, acc1[ct]);
      bf16x16 b2 = b_frag_bf16(w_m1p_bf, DDIM, cb, k, lane);
      acc2[ct] = wmma_bf16(am, b2, acc2[ct]);
    }
  }

  if (tid < 16) { rsum[tid] = 0.0f; rsq[tid] = 0.0f; }
  __syncthreads();

#pragma unroll
  for (int ct = 0; ct < 4; ++ct) {
#pragma unroll
    for (int v = 0; v < 8; ++v) {
      int col = wave * 64 + ct * 16 + (lane & 15);
      int r = v + ((lane >> 4) << 3);
      float a = acc1[ct][v] + b_in[col];
      acc1[ct][v] = a;
      atomicAdd(&rsum[r], a);
      atomicAdd(&rsq[r], a * a);
      float h = acc2[ct][v] + b_m1p[col];
      h = fmaxf(h, 0.0f);
      hm[(size_t)(row0 + r) * HDIM + col] = (__bf16)h;
    }
  }
  __syncthreads();

#pragma unroll
  for (int ct = 0; ct < 4; ++ct) {
#pragma unroll
    for (int v = 0; v < 8; ++v) {
      int col = wave * 64 + ct * 16 + (lane & 15);
      int r = v + ((lane >> 4) << 3);
      float m = rsum[r] * (1.0f / HDIM);
      float var = rsq[r] * (1.0f / HDIM) - m * m;
      float y = (acc1[ct][v] - m) * rsqrtf(var + 1e-5f) * g_ln_in[col] + b_ln_in[col];
      y1[(size_t)(row0 + r) * HDIM + col] = y;
    }
  }
}

// ---------------------------------------------------------------------------
// Kernel B: feats = y1 + 0.5*(hm @ w_m2p^T + b_m2p); K = feats@w_k^T; V = feats@w_v^T
__global__ __launch_bounds__(256) void kB_feats_kv(
    const float* __restrict__ y1, const __bf16* __restrict__ hm,
    const __bf16* __restrict__ w_m2p_bf, const float* __restrict__ b_m2p,
    const __bf16* __restrict__ w_k_bf, const __bf16* __restrict__ w_v_bf,
    float* __restrict__ Kout, float* __restrict__ Vout) {
  __shared__ __bf16 ftile[16 * HDIM];

  const int tid = threadIdx.x;
  const int lane = tid & 31;
  const int wave = tid >> 5;
  const int row0 = blockIdx.x * 16;

  // phase 1: motion layer 2
  f32x8 acc[4];
#pragma unroll
  for (int c = 0; c < 4; ++c) acc[c] = zero8();
  for (int k = 0; k < HDIM; k += 32) {
    bf16x16 a = a_frag(hm + (size_t)row0 * HDIM, HDIM, lane, k);
#pragma unroll
    for (int ct = 0; ct < 4; ++ct) {
      int cb = wave * 64 + ct * 16;
      bf16x16 b = b_frag_bf16(w_m2p_bf, HDIM, cb, k, lane);
      acc[ct] = wmma_bf16(a, b, acc[ct]);
    }
  }
#pragma unroll
  for (int ct = 0; ct < 4; ++ct) {
#pragma unroll
    for (int v = 0; v < 8; ++v) {
      int col = wave * 64 + ct * 16 + (lane & 15);
      int r = v + ((lane >> 4) << 3);
      float f = y1[(size_t)(row0 + r) * HDIM + col] + 0.5f * (acc[ct][v] + b_m2p[col]);
      ftile[r * HDIM + col] = (__bf16)f;
    }
  }
  __syncthreads();

  // phase 2: K and V projections from LDS feats
  f32x8 ka[4], va[4];
#pragma unroll
  for (int c = 0; c < 4; ++c) { ka[c] = zero8(); va[c] = zero8(); }
  for (int k = 0; k < HDIM; k += 32) {
    bf16x16 a = a_frag(ftile, HDIM, lane, k);
#pragma unroll
    for (int ct = 0; ct < 4; ++ct) {
      int cb = wave * 64 + ct * 16;
      bf16x16 bk = b_frag_bf16(w_k_bf, HDIM, cb, k, lane);
      ka[ct] = wmma_bf16(a, bk, ka[ct]);
      bf16x16 bv = b_frag_bf16(w_v_bf, HDIM, cb, k, lane);
      va[ct] = wmma_bf16(a, bv, va[ct]);
    }
  }
#pragma unroll
  for (int ct = 0; ct < 4; ++ct) {
#pragma unroll
    for (int v = 0; v < 8; ++v) {
      int col = wave * 64 + ct * 16 + (lane & 15);
      int r = v + ((lane >> 4) << 3);
      Kout[(size_t)(row0 + r) * HDIM + col] = ka[ct][v];
      Vout[(size_t)(row0 + r) * HDIM + col] = va[ct][v];
    }
  }
}

// ---------------------------------------------------------------------------
// Kernel C: recurrent slot-attention scan, one block per batch element.
__global__ __launch_bounds__(256) void kC_scan(
    const float* __restrict__ slot_noise, const float* __restrict__ slots_mu,
    const float* __restrict__ slots_sigma,
    const float* __restrict__ w_q,
    const float* __restrict__ Kmat, const float* __restrict__ Vmat,
    const float* __restrict__ w_ih, const float* __restrict__ w_hh,
    const float* __restrict__ b_ih, const float* __restrict__ b_hh,
    const float* __restrict__ g_ln_slot, const float* __restrict__ b_ln_slot,
    const float* __restrict__ g_ln_mlp, const float* __restrict__ b_ln_mlp,
    const float* __restrict__ w_mlp1, const float* __restrict__ b_mlp1,
    const float* __restrict__ w_mlp2, const float* __restrict__ b_mlp2,
    float* __restrict__ spf_out, float* __restrict__ att_out) {
  __shared__ float slots[SDIM][HDIM];
  __shared__ float xln[SDIM][HDIM];
  __shared__ float qbuf[SDIM][HDIM];
  __shared__ float upd[SDIM][HDIM];
  __shared__ float hnew[SDIM][HDIM];
  __shared__ float m1[SDIM][2 * HDIM];
  __shared__ float attn[SDIM][NDIM];
  __shared__ float an[SDIM][NDIM];
  __shared__ float stat[SDIM][2];
  __shared__ float asum[SDIM];

  const int b = blockIdx.x;
  const int tid = threadIdx.x;
  const float scale = rsqrtf((float)HDIM);

  for (int i = tid; i < SDIM * HDIM; i += 256) {
    int s = i / HDIM, h = i % HDIM;
    slots[s][h] = slots_mu[i] + slots_sigma[i] * slot_noise[(size_t)b * SDIM * HDIM + i];
  }
  __syncthreads();

  for (int t = 0; t < TDIM; ++t) {
    const float* Kf = Kmat + (size_t)(b * TDIM + t) * NDIM * HDIM;
    const float* Vf = Vmat + (size_t)(b * TDIM + t) * NDIM * HDIM;

    for (int it = 0; it < NITER; ++it) {
      // LN(slots) -> xln
      if (tid < SDIM) {
        float m = 0.0f;
        for (int k = 0; k < HDIM; ++k) m += slots[tid][k];
        m *= (1.0f / HDIM);
        float v = 0.0f;
        for (int k = 0; k < HDIM; ++k) { float d = slots[tid][k] - m; v += d * d; }
        stat[tid][0] = m;
        stat[tid][1] = rsqrtf(v * (1.0f / HDIM) + 1e-5f);
      }
      __syncthreads();
      for (int i = tid; i < SDIM * HDIM; i += 256) {
        int s = i / HDIM, h = i % HDIM;
        xln[s][h] = (slots[s][h] - stat[s][0]) * stat[s][1] * g_ln_slot[h] + b_ln_slot[h];
      }
      __syncthreads();
      // q = xln @ w_q^T
      for (int i = tid; i < SDIM * HDIM; i += 256) {
        int s = i / HDIM, h = i % HDIM;
        const float* w = w_q + (size_t)h * HDIM;
        float a = 0.0f;
        for (int k = 0; k < HDIM; ++k) a += xln[s][k] * w[k];
        qbuf[s][h] = a;
      }
      __syncthreads();
      // logits
      for (int i = tid; i < SDIM * NDIM; i += 256) {
        int s = i / NDIM, n = i % NDIM;
        const float* kr = Kf + (size_t)n * HDIM;
        float a = 0.0f;
        for (int k = 0; k < HDIM; ++k) a += qbuf[s][k] * kr[k];
        attn[s][n] = a * scale;
      }
      __syncthreads();
      // softmax over slots (axis = s)
      for (int n = tid; n < NDIM; n += 256) {
        float mx = fmaxf(attn[0][n], fmaxf(attn[1][n], attn[2][n]));
        float e0 = expf(attn[0][n] - mx);
        float e1 = expf(attn[1][n] - mx);
        float e2 = expf(attn[2][n] - mx);
        float inv = 1.0f / (e0 + e1 + e2);
        attn[0][n] = e0 * inv; attn[1][n] = e1 * inv; attn[2][n] = e2 * inv;
      }
      __syncthreads();
      if (tid < SDIM) {
        float su = 0.0f;
        for (int n = 0; n < NDIM; ++n) su += attn[tid][n];
        asum[tid] = su;
      }
      __syncthreads();
      for (int i = tid; i < SDIM * NDIM; i += 256) {
        int s = i / NDIM, n = i % NDIM;
        an[s][n] = attn[s][n] / (asum[s] + 1e-8f);
      }
      __syncthreads();
      // upd = an @ V
      for (int i = tid; i < SDIM * HDIM; i += 256) {
        int s = i / HDIM, h = i % HDIM;
        float a = 0.0f;
        for (int n = 0; n < NDIM; ++n) a += an[s][n] * Vf[(size_t)n * HDIM + h];
        upd[s][h] = a;
      }
      __syncthreads();
      // GRU cell (gate order r, z, n)
      for (int i = tid; i < SDIM * HDIM; i += 256) {
        int s = i / HDIM, h = i % HDIM;
        float gir = b_ih[h], giz = b_ih[HDIM + h], gin = b_ih[2 * HDIM + h];
        float ghr = b_hh[h], ghz = b_hh[HDIM + h], ghn = b_hh[2 * HDIM + h];
        const float* wir = w_ih + (size_t)h * HDIM;
        const float* wiz = w_ih + (size_t)(HDIM + h) * HDIM;
        const float* win = w_ih + (size_t)(2 * HDIM + h) * HDIM;
        const float* whr = w_hh + (size_t)h * HDIM;
        const float* whz = w_hh + (size_t)(HDIM + h) * HDIM;
        const float* whn = w_hh + (size_t)(2 * HDIM + h) * HDIM;
        for (int k = 0; k < HDIM; ++k) {
          float u = upd[s][k], hh = slots[s][k];
          gir += u * wir[k]; giz += u * wiz[k]; gin += u * win[k];
          ghr += hh * whr[k]; ghz += hh * whz[k]; ghn += hh * whn[k];
        }
        float r = 1.0f / (1.0f + expf(-(gir + ghr)));
        float z = 1.0f / (1.0f + expf(-(giz + ghz)));
        float nn = tanhf(gin + r * ghn);
        hnew[s][h] = (1.0f - z) * nn + z * slots[s][h];
      }
      __syncthreads();
      // LN(hnew) -> xln
      if (tid < SDIM) {
        float m = 0.0f;
        for (int k = 0; k < HDIM; ++k) m += hnew[tid][k];
        m *= (1.0f / HDIM);
        float v = 0.0f;
        for (int k = 0; k < HDIM; ++k) { float d = hnew[tid][k] - m; v += d * d; }
        stat[tid][0] = m;
        stat[tid][1] = rsqrtf(v * (1.0f / HDIM) + 1e-5f);
      }
      __syncthreads();
      for (int i = tid; i < SDIM * HDIM; i += 256) {
        int s = i / HDIM, h = i % HDIM;
        xln[s][h] = (hnew[s][h] - stat[s][0]) * stat[s][1] * g_ln_mlp[h] + b_ln_mlp[h];
      }
      __syncthreads();
      // MLP layer 1 + exact GELU
      for (int i = tid; i < SDIM * 2 * HDIM; i += 256) {
        int s = i / (2 * HDIM), j = i % (2 * HDIM);
        const float* w = w_mlp1 + (size_t)j * HDIM;
        float a = b_mlp1[j];
        for (int k = 0; k < HDIM; ++k) a += xln[s][k] * w[k];
        m1[s][j] = 0.5f * a * (1.0f + erff(a * 0.7071067811865476f));
      }
      __syncthreads();
      // MLP layer 2 + residual
      for (int i = tid; i < SDIM * HDIM; i += 256) {
        int s = i / HDIM, h = i % HDIM;
        const float* w = w_mlp2 + (size_t)h * 2 * HDIM;
        float a = b_mlp2[h];
        for (int k = 0; k < 2 * HDIM; ++k) a += m1[s][k] * w[k];
        slots[s][h] = hnew[s][h] + a;
      }
      __syncthreads();
    }

    // store spf and attn for this frame
    for (int i = tid; i < SDIM * HDIM; i += 256)
      spf_out[(size_t)(b * TDIM + t) * SDIM * HDIM + i] = slots[i / HDIM][i % HDIM];
    for (int i = tid; i < SDIM * NDIM; i += 256)
      att_out[(size_t)(b * TDIM + t) * SDIM * NDIM + i] = attn[i / NDIM][i % NDIM];
    __syncthreads();
  }
}

// ---------------------------------------------------------------------------
// Kernel D1: temporal attention, one block per (track, head)
__global__ __launch_bounds__(256) void kD1_attn(
    const float* __restrict__ spf,
    const float* __restrict__ w_tq, const float* __restrict__ b_tq,
    const float* __restrict__ w_tk, const float* __restrict__ b_tk,
    const float* __restrict__ w_tv, const float* __restrict__ b_tv,
    float* __restrict__ o_ws) {
  __shared__ float x[TDIM][HDIM];
  __shared__ float qh[TDIM][HD];
  __shared__ float kh[TDIM][HD];
  __shared__ float vh[TDIM][HD];
  __shared__ float sc[TDIM][TDIM];

  const int tid = threadIdx.x;
  const int track = blockIdx.x >> 3;   // b*3+s
  const int head = blockIdx.x & 7;
  const int b = track / SDIM, s = track % SDIM;

  for (int i = tid; i < TDIM * HDIM; i += 256) {
    int t = i / HDIM, h = i % HDIM;
    x[t][h] = spf[(size_t)((b * TDIM + t) * SDIM + s) * HDIM + h];
  }
  __syncthreads();

  for (int i = tid; i < 3 * TDIM * HD; i += 256) {
    int m = i / (TDIM * HD);
    int r = i % (TDIM * HD);
    int t = r / HD, d = r % HD;
    int row = head * HD + d;
    const float* w = (m == 0 ? w_tq : (m == 1 ? w_tk : w_tv)) + (size_t)row * HDIM;
    float a = (m == 0 ? b_tq : (m == 1 ? b_tk : b_tv))[row];
    for (int k = 0; k < HDIM; ++k) a += x[t][k] * w[k];
    if (m == 0) qh[t][d] = a; else if (m == 1) kh[t][d] = a; else vh[t][d] = a;
  }
  __syncthreads();

  for (int i = tid; i < TDIM * TDIM; i += 256) {
    int t1 = i / TDIM, t2 = i % TDIM;
    float a = 0.0f;
    for (int d = 0; d < HD; ++d) a += qh[t1][d] * kh[t2][d];
    sc[t1][t2] = a * 0.125f;  // 1/sqrt(64)
  }
  __syncthreads();
  if (tid < TDIM) {
    float mx = -1e30f;
    for (int t2 = 0; t2 < TDIM; ++t2) mx = fmaxf(mx, sc[tid][t2]);
    float su = 0.0f;
    for (int t2 = 0; t2 < TDIM; ++t2) { float e = expf(sc[tid][t2] - mx); sc[tid][t2] = e; su += e; }
    float inv = 1.0f / su;
    for (int t2 = 0; t2 < TDIM; ++t2) sc[tid][t2] *= inv;
  }
  __syncthreads();
  for (int i = tid; i < TDIM * HD; i += 256) {
    int t = i / HD, d = i % HD;
    float a = 0.0f;
    for (int t2 = 0; t2 < TDIM; ++t2) a += sc[t][t2] * vh[t2][d];
    o_ws[(size_t)(track * TDIM + t) * HDIM + head * HD + d] = a;
  }
}

// Kernel D2: out-proj + residual + LN + mean over T -> slots_agg
__global__ __launch_bounds__(256) void kD2_out(
    const float* __restrict__ spf, const float* __restrict__ o_ws,
    const float* __restrict__ w_to, const float* __restrict__ b_to,
    const float* __restrict__ g_ln_t, const float* __restrict__ b_ln_t,
    float* __restrict__ agg_out) {
  __shared__ float xb[TDIM][HDIM];
  __shared__ float rm[TDIM], rr[TDIM];

  const int tid = threadIdx.x;
  const int track = blockIdx.x;
  const int b = track / SDIM, s = track % SDIM;

  for (int i = tid; i < TDIM * HDIM; i += 256) {
    int t = i / HDIM, h = i % HDIM;
    xb[t][h] = spf[(size_t)((b * TDIM + t) * SDIM + s) * HDIM + h];
  }
  __syncthreads();

  for (int i = tid; i < TDIM * HDIM; i += 256) {
    int t = i / HDIM, h = i % HDIM;
    const float* orow = o_ws + (size_t)(track * TDIM + t) * HDIM;
    const float* w = w_to + (size_t)h * HDIM;
    float a = b_to[h];
    for (int k = 0; k < HDIM; ++k) a += orow[k] * w[k];
    xb[t][h] = a + xb[t][h];   // each (t,h) owned by exactly one thread
  }
  __syncthreads();

  if (tid < TDIM) {
    float m = 0.0f;
    for (int k = 0; k < HDIM; ++k) m += xb[tid][k];
    m *= (1.0f / HDIM);
    float v = 0.0f;
    for (int k = 0; k < HDIM; ++k) { float d = xb[tid][k] - m; v += d * d; }
    rm[tid] = m;
    rr[tid] = rsqrtf(v * (1.0f / HDIM) + 1e-5f);
  }
  __syncthreads();

  for (int h = tid; h < HDIM; h += 256) {
    float su = 0.0f;
    for (int t = 0; t < TDIM; ++t)
      su += (xb[t][h] - rm[t]) * rr[t] * g_ln_t[h] + b_ln_t[h];
    agg_out[(size_t)track * HDIM + h] = su * (1.0f / TDIM);
  }
}

// ---------------------------------------------------------------------------
extern "C" void kernel_launch(void* const* d_in, const int* in_sizes, int n_in,
                              void* d_out, int out_size, void* d_ws, size_t ws_size,
                              hipStream_t stream) {
  (void)in_sizes; (void)n_in; (void)out_size; (void)ws_size;
  const float* X          = (const float*)d_in[0];
  const float* slot_noise = (const float*)d_in[1];
  const float* slots_mu   = (const float*)d_in[2];
  const float* slots_sig  = (const float*)d_in[3];
  const float* w_in  = (const float*)d_in[4];
  const float* b_in  = (const float*)d_in[5];
  const float* g_ln_in = (const float*)d_in[6];
  const float* b_ln_in = (const float*)d_in[7];
  const float* w_m1p = (const float*)d_in[8];
  const float* b_m1p = (const float*)d_in[9];
  const float* w_m2p = (const float*)d_in[10];
  const float* b_m2p = (const float*)d_in[11];
  const float* w_q = (const float*)d_in[12];
  const float* w_k = (const float*)d_in[13];
  const float* w_v = (const float*)d_in[14];
  const float* w_ih = (const float*)d_in[15];
  const float* w_hh = (const float*)d_in[16];
  const float* b_ih = (const float*)d_in[17];
  const float* b_hh = (const float*)d_in[18];
  const float* g_ln_slot = (const float*)d_in[19];
  const float* b_ln_slot = (const float*)d_in[20];
  const float* g_ln_mlp = (const float*)d_in[21];
  const float* b_ln_mlp = (const float*)d_in[22];
  const float* w_mlp1 = (const float*)d_in[23];
  const float* b_mlp1 = (const float*)d_in[24];
  const float* w_mlp2 = (const float*)d_in[25];
  const float* b_mlp2 = (const float*)d_in[26];
  const float* w_tq = (const float*)d_in[27];
  const float* b_tq = (const float*)d_in[28];
  const float* w_tk = (const float*)d_in[29];
  const float* b_tk = (const float*)d_in[30];
  const float* w_tv = (const float*)d_in[31];
  const float* b_tv = (const float*)d_in[32];
  const float* w_to = (const float*)d_in[33];
  const float* b_to = (const float*)d_in[34];
  const float* g_ln_t = (const float*)d_in[35];
  const float* b_ln_t = (const float*)d_in[36];

  // workspace layout
  char* ws = (char*)d_ws;
  float*  y1   = (float*)ws;                                    // 32768*512 f32
  __bf16* hm   = (__bf16*)(ws + (size_t)ROWS * HDIM * 4);       // 32768*512 bf16
  float*  Kmat = (float*)(ws + (size_t)ROWS * HDIM * 6);        // 32768*512 f32
  float*  Vmat = (float*)(ws + (size_t)ROWS * HDIM * 10);       // 32768*512 f32
  float*  o_ws = (float*)(ws + (size_t)ROWS * HDIM * 14);       // 24*16*512 f32
  char*   wb   = ws + (size_t)ROWS * HDIM * 14 + (1 << 20);     // bf16 weight arena
  __bf16* w_in_bf  = (__bf16*)(wb);                             // 512*1280
  __bf16* w_m1p_bf = (__bf16*)(wb + 1310720);                   // 512*1280
  __bf16* w_m2p_bf = (__bf16*)(wb + 2621440);                   // 512*512
  __bf16* w_k_bf   = (__bf16*)(wb + 3145728);                   // 512*512
  __bf16* w_v_bf   = (__bf16*)(wb + 3670016);                   // 512*512

  float* out = (float*)d_out;
  float* agg_out = out;                                   // [B,S,H]   12288
  float* spf_out = out + BDIM * SDIM * HDIM;              // [B,T,S,H] 196608
  float* att_out = spf_out + BDIM * TDIM * SDIM * HDIM;   // [B,T,S,N] 98304

  // one-time weight conversion to bf16 (lives in L2 for the GEMM kernels)
  kW_cvt<<<(HDIM * DDIM + 255) / 256, 256, 0, stream>>>(w_in,  w_in_bf,  HDIM * DDIM);
  kW_cvt<<<(HDIM * DDIM + 255) / 256, 256, 0, stream>>>(w_m1p, w_m1p_bf, HDIM * DDIM);
  kW_cvt<<<(HDIM * HDIM + 255) / 256, 256, 0, stream>>>(w_m2p, w_m2p_bf, HDIM * HDIM);
  kW_cvt<<<(HDIM * HDIM + 255) / 256, 256, 0, stream>>>(w_k,   w_k_bf,   HDIM * HDIM);
  kW_cvt<<<(HDIM * HDIM + 255) / 256, 256, 0, stream>>>(w_v,   w_v_bf,   HDIM * HDIM);

  kA_proj<<<ROWS / 16, 256, 0, stream>>>(X, w_in_bf, b_in, g_ln_in, b_ln_in,
                                         w_m1p_bf, b_m1p, y1, hm);
  kB_feats_kv<<<ROWS / 16, 256, 0, stream>>>(y1, hm, w_m2p_bf, b_m2p, w_k_bf, w_v_bf,
                                             Kmat, Vmat);
  kC_scan<<<BDIM, 256, 0, stream>>>(slot_noise, slots_mu, slots_sig, w_q,
                                    Kmat, Vmat, w_ih, w_hh, b_ih, b_hh,
                                    g_ln_slot, b_ln_slot, g_ln_mlp, b_ln_mlp,
                                    w_mlp1, b_mlp1, w_mlp2, b_mlp2,
                                    spf_out, att_out);
  kD1_attn<<<BDIM * SDIM * NHEADS, 256, 0, stream>>>(spf_out, w_tq, b_tq, w_tk,
                                                     b_tk, w_tv, b_tv, o_ws);
  kD2_out<<<BDIM * SDIM, 256, 0, stream>>>(spf_out, o_ws, w_to, b_to,
                                           g_ln_t, b_ln_t, agg_out);
}
